// ReservoirCell_14989435863542
// MI455X (gfx1250) — matrix operand order repaired and checked by reading, not compiled, not run on hardware
//
#include <hip/hip_runtime.h>

// Problem constants (match reference)
#define B_ 128
#define T_ 512
#define I_ 128
#define R_ 1024

#define HSTRIDE 1032  // 1024 + 8 bf16 pad -> 2064B row stride, bank-conflict-free
#define XSTRIDE 136   // 128 + 8 bf16 pad

typedef __attribute__((ext_vector_type(16))) __bf16 v16bf;
typedef __attribute__((ext_vector_type(8)))  __bf16 bf16x8;
typedef __attribute__((ext_vector_type(8)))  float  v8f;

__device__ __forceinline__ unsigned short f32_to_bf16_rne(float f) {
  unsigned int u = __float_as_uint(f);
  u += 0x7FFFu + ((u >> 16) & 1u);
  return (unsigned short)(u >> 16);
}

// Build the 16-element bf16 A/B operand from two 16-byte contiguous runs.
__device__ __forceinline__ v16bf make16(const unsigned short* p0, const unsigned short* p1) {
  bf16x8 lo = *reinterpret_cast<const bf16x8*>(p0);
  bf16x8 hi = *reinterpret_cast<const bf16x8*>(p1);
  v16bf r;
#pragma unroll
  for (int i = 0; i < 8; ++i) { r[i] = lo[i]; r[i + 8] = hi[i]; }
  return r;
}

// ---- Setup kernels ----

__global__ void k_transpose_bf16(const float* __restrict__ src,
                                 unsigned short* __restrict__ dst,
                                 int K, int N) {
  int i = blockIdx.x * blockDim.x + threadIdx.x;
  if (i < K * N) {
    int n = i / K;
    int k = i - n * K;
    dst[i] = f32_to_bf16_rne(src[(size_t)k * N + n]);
  }
}

__global__ void k_cvt_bf16(const float* __restrict__ src,
                           unsigned short* __restrict__ dst, int n) {
  int i = blockIdx.x * blockDim.x + threadIdx.x;
  if (i < n) dst[i] = f32_to_bf16_rne(src[i]);
}

__global__ void k_init(unsigned short* __restrict__ hbf, float* __restrict__ hfp,
                       unsigned int* __restrict__ bar, int n) {
  int i = blockIdx.x * blockDim.x + threadIdx.x;
  if (i < n) { hbf[i] = 0; hfp[i] = 0.0f; }
  if (i < 2) bar[i] = 0u;
}

// ---- Device-wide sense/generation barrier ----
__device__ __forceinline__ void grid_sync(unsigned int* cnt, unsigned int* gen,
                                          unsigned int nblocks) {
  __syncthreads();
  if (threadIdx.x == 0) {
    __threadfence();
    unsigned int g = __hip_atomic_load(gen, __ATOMIC_RELAXED, __HIP_MEMORY_SCOPE_AGENT);
    unsigned int arrived =
        __hip_atomic_fetch_add(cnt, 1u, __ATOMIC_ACQ_REL, __HIP_MEMORY_SCOPE_AGENT);
    if (arrived == nblocks - 1u) {
      __hip_atomic_store(cnt, 0u, __ATOMIC_RELAXED, __HIP_MEMORY_SCOPE_AGENT);
      __hip_atomic_store(gen, g + 1u, __ATOMIC_RELEASE, __HIP_MEMORY_SCOPE_AGENT);
    } else {
      while (__hip_atomic_load(gen, __ATOMIC_ACQUIRE, __HIP_MEMORY_SCOPE_AGENT) == g) {
        __builtin_amdgcn_s_sleep(2);
      }
    }
    __threadfence();
  }
  __syncthreads();
}

// ---- Persistent full-scan kernel: whole T loop in one launch ----
__global__ __launch_bounds__(128) void k_scan(
    const unsigned short* __restrict__ xbf,    // [B,T,I] bf16
    const unsigned short* __restrict__ winT,   // [R,I]  bf16 (N-major)
    const unsigned short* __restrict__ wrecT,  // [R,R]  bf16 (N-major)
    const float* __restrict__ bias,            // [R] f32
    unsigned short* __restrict__ hbf,          // [2][B,R] bf16
    float* __restrict__ hfp,                   // [2][B,R] f32
    float* __restrict__ out,                   // [B,T,R] f32
    unsigned int* __restrict__ bar,            // {cnt, gen}
    int nblocks) {
  __shared__ __align__(16) unsigned short sH[16 * HSTRIDE];
  __shared__ __align__(16) unsigned short sX[16 * XSTRIDE];

  const int tid  = threadIdx.x;
  const int lane = tid & 31;
  const int wave = tid >> 5;
  const int totalWaves = nblocks * 4;
  const int maxTrips = (512 + totalWaves - 1) / totalWaves;
  const int half = lane >> 4;
  const int lcol = lane & 15;
  const int BR = B_ * R_;

  for (int t = 0; t < T_; ++t) {
    const unsigned short* hprev_bf = hbf + (size_t)(t & 1) * BR;
    const float*          hprev_fp = hfp + (size_t)(t & 1) * BR;
    unsigned short*       hcur_bf  = hbf + (size_t)((t + 1) & 1) * BR;
    float*                hcur_fp  = hfp + (size_t)((t + 1) & 1) * BR;

    for (int trip = 0; trip < maxTrips; ++trip) {
      const int wBlock0 = blockIdx.x * 4 + trip * totalWaves;  // block-uniform
      if (wBlock0 < 512) {
        // 4 consecutive wave-tiles never cross a 64-tile N-group => same m tile
        const int mB = wBlock0 >> 6;
        // ---- stage A rows (h_{t-1} and x_t, 16 batch rows) into LDS ----
        {
          const unsigned short* hs = hprev_bf + (size_t)(mB * 16) * R_;
          for (int i = tid; i < 2048; i += 128) {  // 16 rows x 1024 / 8
            int row = i >> 7, off = (i & 127) * 8;
            *reinterpret_cast<uint4*>(sH + row * HSTRIDE + off) =
                *reinterpret_cast<const uint4*>(hs + (size_t)row * R_ + off);
          }
          const unsigned short* xs = xbf + (size_t)(mB * 16) * (T_ * I_) + (size_t)t * I_;
          for (int i = tid; i < 256; i += 128) {   // 16 rows x 128 / 8
            int row = i >> 4, off = (i & 15) * 8;
            *reinterpret_cast<uint4*>(sX + row * XSTRIDE + off) =
                *reinterpret_cast<const uint4*>(xs + (size_t)row * (T_ * I_) + off);
          }
        }
        __syncthreads();

        const int w = wBlock0 + wave;
        if (w < 512) {
          const int n = w & 63;
          const int n0 = n * 16;
          const int m0 = (w >> 6) * 16;

          v8f c = {};

          // ---- x_t @ Win (K = 128), A from LDS ----
          {
            const unsigned short* aRow = sX + lcol * XSTRIDE;
            const unsigned short* bRow = winT + (size_t)(n0 + lcol) * I_;
#pragma unroll
            for (int kk = 0; kk < I_; kk += 32) {
              v16bf a = make16(aRow + kk + half * 8, aRow + kk + 16 + half * 8);
              v16bf b = make16(bRow + kk + half * 8, bRow + kk + 16 + half * 8);
              c = __builtin_amdgcn_wmma_f32_16x16x32_bf16(false, a, false, b,
                                                          (short)0, c, false, false);
            }
          }
          // ---- h_{t-1} @ Wrec (K = 1024), A from LDS ----
          {
            const unsigned short* aRow = sH + lcol * HSTRIDE;
            const unsigned short* bRow = wrecT + (size_t)(n0 + lcol) * R_;
#pragma unroll 4
            for (int kk = 0; kk < R_; kk += 32) {
              v16bf a = make16(aRow + kk + half * 8, aRow + kk + 16 + half * 8);
              v16bf b = make16(bRow + kk + half * 8, bRow + kk + 16 + half * 8);
              c = __builtin_amdgcn_wmma_f32_16x16x32_bf16(false, a, false, b,
                                                          (short)0, c, false, false);
            }
          }
          // ---- bias + tanh + leaky integration (f32) ----
          const int col = n0 + lcol;
          const float bv = bias[col];
#pragma unroll
          for (int r = 0; r < 8; ++r) {
            const int row = m0 + half * 8 + r;
            const float hp = hprev_fp[(size_t)row * R_ + col];
            const float pre = c[r] + bv;
            const float h = 0.1f * hp + 0.9f * tanhf(pre);
            out[(size_t)row * (T_ * R_) + (size_t)t * R_ + col] = h;
            hcur_fp[(size_t)row * R_ + col] = h;
            hcur_bf[(size_t)row * R_ + col] = f32_to_bf16_rne(h);
          }
        }
        __syncthreads();
      }
    }
    grid_sync(bar, bar + 1, (unsigned int)nblocks);
  }
}

// ---- Fallback: one launch per step (no LDS staging) ----
__global__ __launch_bounds__(128) void k_step(
    const unsigned short* __restrict__ xbf, const unsigned short* __restrict__ winT,
    const unsigned short* __restrict__ wrecT, const float* __restrict__ bias,
    const unsigned short* __restrict__ hprev_bf, const float* __restrict__ hprev_fp,
    unsigned short* __restrict__ hcur_bf, float* __restrict__ hcur_fp,
    float* __restrict__ out, int t) {
  const int lane = threadIdx.x & 31;
  const int wave = threadIdx.x >> 5;
  const int w = blockIdx.x * 4 + wave;
  const int n = w & 63, m = w >> 6;
  const int half = lane >> 4, lcol = lane & 15;
  const int m0 = m * 16, n0 = n * 16;
  v8f c = {};
  {
    const unsigned short* aRow = xbf + (size_t)(m0 + lcol) * (T_ * I_) + (size_t)t * I_;
    const unsigned short* bRow = winT + (size_t)(n0 + lcol) * I_;
#pragma unroll
    for (int kk = 0; kk < I_; kk += 32) {
      v16bf a = make16(aRow + kk + half * 8, aRow + kk + 16 + half * 8);
      v16bf b = make16(bRow + kk + half * 8, bRow + kk + 16 + half * 8);
      c = __builtin_amdgcn_wmma_f32_16x16x32_bf16(false, a, false, b, (short)0, c, false, false);
    }
  }
  {
    const unsigned short* aRow = hprev_bf + (size_t)(m0 + lcol) * R_;
    const unsigned short* bRow = wrecT + (size_t)(n0 + lcol) * R_;
#pragma unroll 4
    for (int kk = 0; kk < R_; kk += 32) {
      v16bf a = make16(aRow + kk + half * 8, aRow + kk + 16 + half * 8);
      v16bf b = make16(bRow + kk + half * 8, bRow + kk + 16 + half * 8);
      c = __builtin_amdgcn_wmma_f32_16x16x32_bf16(false, a, false, b, (short)0, c, false, false);
    }
  }
  const int col = n0 + lcol;
  const float bv = bias[col];
#pragma unroll
  for (int r = 0; r < 8; ++r) {
    const int row = m0 + half * 8 + r;
    const float hp = hprev_fp[(size_t)row * R_ + col];
    const float h = 0.1f * hp + 0.9f * tanhf(c[r] + bv);
    out[(size_t)row * (T_ * R_) + (size_t)t * R_ + col] = h;
    hcur_fp[(size_t)row * R_ + col] = h;
    hcur_bf[(size_t)row * R_ + col] = f32_to_bf16_rne(h);
  }
}

extern "C" void kernel_launch(void* const* d_in, const int* in_sizes, int n_in,
                              void* d_out, int out_size, void* d_ws, size_t ws_size,
                              hipStream_t stream) {
  const float* x    = (const float*)d_in[0];  // [B,T,I]
  const float* win  = (const float*)d_in[1];  // [I,R]
  const float* wrec = (const float*)d_in[2];  // [R,R]
  const float* bias = (const float*)d_in[3];  // [R]
  float* out = (float*)d_out;                 // [B,T,R]

  char* ws = (char*)d_ws;
  unsigned short* winT  = (unsigned short*)(ws + 0);          //  256 KB
  unsigned short* wrecT = (unsigned short*)(ws + 262144);     //    2 MB
  unsigned short* xbf   = (unsigned short*)(ws + 2359296);    //   16 MB
  unsigned short* hbf   = (unsigned short*)(ws + 19136512);   // 2x256 KB
  float*          hfp   = (float*)        (ws + 19660800);    // 2x512 KB
  unsigned int*   bar   = (unsigned int*) (ws + 20709376);    // {cnt, gen}

  const int BR = B_ * R_;

  k_transpose_bf16<<<(I_ * R_ + 255) / 256, 256, 0, stream>>>(win, winT, I_, R_);
  k_transpose_bf16<<<(R_ * R_ + 255) / 256, 256, 0, stream>>>(wrec, wrecT, R_, R_);
  k_cvt_bf16<<<(B_ * T_ * I_ + 255) / 256, 256, 0, stream>>>(x, xbf, B_ * T_ * I_);
  k_init<<<(BR + 255) / 256, 256, 0, stream>>>(hbf, hfp, bar, BR);

  // Persistent grid sizing: one small block per CU is always co-resident.
  int dev = 0;
  (void)hipGetDevice(&dev);
  int numCU = 0;
  if (hipDeviceGetAttribute(&numCU, hipDeviceAttributeMultiprocessorCount, dev) != hipSuccess)
    numCU = 0;

  if (numCU >= 1) {
    const int grid = numCU < 128 ? numCU : 128;
    k_scan<<<grid, 128, 0, stream>>>(xbf, winT, wrecT, bias, hbf, hfp, out, bar, grid);
  } else {
    for (int t = 0; t < T_; ++t) {
      const int cur = t & 1, nxt = (t + 1) & 1;
      k_step<<<128, 128, 0, stream>>>(xbf, winT, wrecT, bias,
                                      hbf + (size_t)cur * BR, hfp + (size_t)cur * BR,
                                      hbf + (size_t)nxt * BR, hfp + (size_t)nxt * BR,
                                      out, t);
    }
  }
}
